// PointCNN_29789893165481
// MI455X (gfx1250) — matrix operand order repaired and checked by reading, not compiled, not run on hardware
//
#include <hip/hip_runtime.h>
#include <hip/hip_bf16.h>
#include <math.h>

typedef __attribute__((ext_vector_type(16))) _Float16 v16h;
typedef __attribute__((ext_vector_type(8)))  float    v8f;

static inline int cdiv(int a, int b) { return (a + b - 1) / b; }

// ---------------------------------------------------------------------------
// kNN: per point, keep KK nearest (ascending dist, lower index wins ties,
// matching jax.lax.top_k(-d)), then emit every dil-th -> K indices.
// ---------------------------------------------------------------------------
template <int KK>
__global__ void knn_kernel(const float* __restrict__ pos, int B, int n,
                           int dil, int K, int* __restrict__ nbr) {
  int t = blockIdx.x * blockDim.x + threadIdx.x;
  if (t >= B * n) return;
  int b = t / n, i = t % n;
  const float* p = pos + (size_t)b * n * 3;
  float xi = p[i * 3 + 0], yi = p[i * 3 + 1], zi = p[i * 3 + 2];
  float sqi = xi * xi + yi * yi + zi * zi;
  float bd[KK];
  int   bi[KK];
#pragma unroll
  for (int m = 0; m < KK; ++m) { bd[m] = 3.4e38f; bi[m] = 0; }
  for (int j = 0; j < n; ++j) {
    float xj = p[j * 3 + 0], yj = p[j * 3 + 1], zj = p[j * 3 + 2];
    float d = sqi + (xj * xj + yj * yj + zj * zj)
            - 2.f * (xi * xj + yi * yj + zi * zj);
    if (d < bd[KK - 1]) {
      int m = KK - 1;
      while (m > 0 && d < bd[m - 1]) { bd[m] = bd[m - 1]; bi[m] = bi[m - 1]; --m; }
      bd[m] = d; bi[m] = j;
    }
  }
  int* o = nbr + (size_t)t * K;
  for (int k = 0; k < K; ++k) o[k] = bi[k * dil];
}

// ---------------------------------------------------------------------------
// Farthest point sampling: one block per cloud; min-dist array in LDS,
// block argmax (first-max tie break like jnp.argmax), start at index 0.
// ---------------------------------------------------------------------------
__global__ void fps_kernel(const float* __restrict__ pos, int n, int nsel,
                           int* __restrict__ sel) {
  __shared__ float mind[1024];
  __shared__ float rv[256];
  __shared__ int   ri[256];
  __shared__ float curp[3];
  __shared__ int   cur;
  int b = blockIdx.x, tid = threadIdx.x;
  const float* p = pos + (size_t)b * n * 3;
  int* s = sel + (size_t)b * nsel;
  for (int i = tid; i < n; i += blockDim.x) mind[i] = 1e10f;
  if (tid == 0) { s[0] = 0; cur = 0; }
  __syncthreads();
  for (int it = 1; it < nsel; ++it) {
    if (tid < 3) curp[tid] = p[cur * 3 + tid];
    __syncthreads();
    float cx = curp[0], cy = curp[1], cz = curp[2];
    float bv = -1.f; int bidx = n;
    for (int i = tid; i < n; i += blockDim.x) {
      float dx = p[i * 3 + 0] - cx, dy = p[i * 3 + 1] - cy, dz = p[i * 3 + 2] - cz;
      float m = fminf(mind[i], dx * dx + dy * dy + dz * dz);
      mind[i] = m;
      if (m > bv) { bv = m; bidx = i; }
    }
    rv[tid] = bv; ri[tid] = bidx;
    __syncthreads();
    for (int st = blockDim.x >> 1; st > 0; st >>= 1) {
      if (tid < st) {
        if (rv[tid + st] > rv[tid] ||
            (rv[tid + st] == rv[tid] && ri[tid + st] < ri[tid])) {
          rv[tid] = rv[tid + st]; ri[tid] = ri[tid + st];
        }
      }
      __syncthreads();
    }
    if (tid == 0) { s[it] = ri[0]; cur = ri[0]; }
    __syncthreads();
  }
}

// ---------------------------------------------------------------------------
// Per-cloud gather: out[b,r,c] = in[b, idx[b,r], c]
// ---------------------------------------------------------------------------
__global__ void gather_kernel(const float* __restrict__ in, const int* __restrict__ idx,
                              float* __restrict__ out, int B, int nout, int nin, int C) {
  int t = blockIdx.x * blockDim.x + threadIdx.x;
  if (t >= B * nout * C) return;
  int c = t % C, r = (t / C) % nout, b = t / (C * nout);
  out[t] = in[((size_t)b * nin + idx[b * nout + r]) * C + c];
}

// rel[b,i,k,:] = pos[b,nbr,:] - pos[b,i,:]
__global__ void rel_kernel(const float* __restrict__ pos, const int* __restrict__ nbr,
                           float* __restrict__ rel, int B, int n, int K) {
  int t = blockIdx.x * blockDim.x + threadIdx.x;
  if (t >= B * n * K * 3) return;
  int d = t % 3, k = (t / 3) % K, pI = t / (3 * K);
  int b = pI / n;
  int j = nbr[(size_t)pI * K + k];
  rel[t] = pos[((size_t)b * n + j) * 3 + d] - pos[(size_t)pI * 3 + d];
}

// Gather previous features into the trailing ci columns of x_star (row stride C)
__global__ void gatherx_kernel(const float* __restrict__ x, const int* __restrict__ nbr,
                               float* __restrict__ xs, int B, int n, int K,
                               int ci, int C, int cd) {
  int t = blockIdx.x * blockDim.x + threadIdx.x;
  if (t >= B * n * K * ci) return;
  int c = t % ci, k = (t / ci) % K, pI = t / (ci * K);
  int b = pI / n;
  xs[((size_t)pI * K + k) * C + cd + c] =
      x[((size_t)b * n + nbr[(size_t)pI * K + k]) * ci + c];
}

// ---------------------------------------------------------------------------
// Stage f32 operands into zero-padded f16 WMMA-fragment layout.
// A 16x32 fragment: lane L (0-15) holds row tm*16+L, halves e: K = hlf*8+e+(e>=8?8:0)
// B 32x16 fragment: lane L holds col tn*16+(L&15),     halves e: K = hlf*8+e+(e>=8?8:0)
// Storage: fragment f = (tile*nK + kc); element ((f*32 + lane)*16 + e)
// ---------------------------------------------------------------------------
__global__ void stage_a_kernel(const float* __restrict__ A, int lda, int M, int Kd,
                               int nK, int total, _Float16* __restrict__ out) {
  int t = blockIdx.x * blockDim.x + threadIdx.x;
  if (t >= total) return;
  int e = t & 15, lane = (t >> 4) & 31, frag = t >> 9;
  int kc = frag % nK, tm = frag / nK;
  int row = tm * 16 + (lane & 15);
  int kk = kc * 32 + (lane >> 4) * 8 + e + ((e >= 8) ? 8 : 0);
  out[t] = (_Float16)((row < M && kk < Kd) ? A[(size_t)row * lda + kk] : 0.f);
}

__global__ void stage_w_kernel(const float* __restrict__ W, int Nout, int Kd,
                               int nK, int total, _Float16* __restrict__ out) {
  int t = blockIdx.x * blockDim.x + threadIdx.x;
  if (t >= total) return;
  int e = t & 15, lane = (t >> 4) & 31, frag = t >> 9;
  int kc = frag % nK, tn = frag / nK;
  int col = tn * 16 + (lane & 15);
  int kk = kc * 32 + (lane >> 4) * 8 + e + ((e >= 8) ? 8 : 0);
  out[t] = (_Float16)((col < Nout && kk < Kd) ? W[(size_t)col * Kd + kk] : 0.f);
}

// ---------------------------------------------------------------------------
// WMMA GEMM on pre-staged fragments: C = act(A @ W^T + bias).
// One wave computes a 16x32 output slab: one A fragment feeds two independent
// accumulators (no D->C RAW hazard between the paired v_wmma). nK is always
// EVEN (host rounds up; pad chunks are staged as zeros), so the k-loop
// processes an E/O chunk pair per iteration with all four v_wmma
// unconditional -- no mid-loop exit, so accumulator phis coalesce and no
// copies are generated. Fragment loads for the O chunk and the next E chunk
// are issued before the WMMAs that consume the current chunks, giving
// partial s_wait_loadcnt overlap. For an odd N-tile count the second B
// fragment reads stale staging data whose results are never stored
// (epilogue guards col < Nout); WMMA raises no exceptions (ISA 7.12).
// act: 0=none, 1=ELU, 2=ReLU.
// ---------------------------------------------------------------------------
__global__ void gemm_wmma(const _Float16* __restrict__ Af,
                          const _Float16* __restrict__ Wf,
                          const float* __restrict__ bias,
                          float* __restrict__ C, int ldc,
                          int M, int Nout, int nK, int act) {
  int tilesM = (M + 15) >> 4;
  int tilesN = (Nout + 15) >> 4;
  int tilesN2 = (tilesN + 1) >> 1;
  int wave = blockIdx.x * (blockDim.x >> 5) + (threadIdx.x >> 5);
  if (wave >= tilesM * tilesN2) return;
  int tm = wave / tilesN2;
  int tn0 = (wave % tilesN2) * 2;
  int lane = threadIdx.x & 31;
  int hlf = lane >> 4, l16 = lane & 15;

  const v16h* Av = (const v16h*)Af + (size_t)tm * nK * 32 + lane;
  const v16h* B0 = (const v16h*)Wf + (size_t)tn0 * nK * 32 + lane;
  const v16h* B1 = B0 + (size_t)nK * 32;

  union { v8f v; float f[8]; } acc0, acc1;
#pragma unroll
  for (int r = 0; r < 8; ++r) { acc0.f[r] = 0.f; acc1.f[r] = 0.f; }

  // E stage holds chunk kc; O stage holds chunk kc+1 (always valid: nK even)
  v16h aE = Av[0], pE = B0[0], qE = B1[0];
  for (int kc = 0; kc < nK; kc += 2) {
    v16h aO = Av[(size_t)(kc + 1) * 32];
    v16h pO = B0[(size_t)(kc + 1) * 32];
    v16h qO = B1[(size_t)(kc + 1) * 32];
    // speculative prefetch two chunks ahead (dropped on fault per ISA)
    __builtin_prefetch((const void*)(Av + (size_t)(kc + 3) * 32), 0, 1);
    __builtin_prefetch((const void*)(B0 + (size_t)(kc + 3) * 32), 0, 1);
    acc0.v = __builtin_amdgcn_wmma_f32_16x16x32_f16(
        false, aE, false, pE, (short)0, acc0.v, false, false);
    acc1.v = __builtin_amdgcn_wmma_f32_16x16x32_f16(
        false, aE, false, qE, (short)0, acc1.v, false, false);
    if (kc + 2 < nK) {               // issue loads for next E before using O
      aE = Av[(size_t)(kc + 2) * 32];
      pE = B0[(size_t)(kc + 2) * 32];
      qE = B1[(size_t)(kc + 2) * 32];
    }
    acc0.v = __builtin_amdgcn_wmma_f32_16x16x32_f16(
        false, aO, false, pO, (short)0, acc0.v, false, false);
    acc1.v = __builtin_amdgcn_wmma_f32_16x16x32_f16(
        false, aO, false, qO, (short)0, acc1.v, false, false);
  }

#pragma unroll 2
  for (int half = 0; half < 2; ++half) {
    int col = (tn0 + half) * 16 + l16;
    if (col >= Nout) continue;
    float bv = bias ? bias[col] : 0.f;
    const float* accp = half ? acc1.f : acc0.f;
#pragma unroll
    for (int r = 0; r < 8; ++r) {
      int rr = tm * 16 + hlf * 8 + r;
      if (rr < M) {
        float v = accp[r] + bv;
        if (act == 1)      v = (v > 0.f) ? v : (__expf(v) - 1.f);
        else if (act == 2) v = fmaxf(v, 0.f);
        C[(size_t)rr * ldc + col] = v;
      }
    }
  }
}

// ---------------------------------------------------------------------------
// BatchNorm (training): per-column mean/var over M rows -> scale/shift
// ---------------------------------------------------------------------------
__global__ void bnstats_kernel(const float* __restrict__ X, int M, int ld,
                               const float* __restrict__ g, const float* __restrict__ b,
                               float* __restrict__ scale, float* __restrict__ shift) {
  __shared__ float s1[256], s2[256];
  int col = blockIdx.x;
  float a = 0.f, q = 0.f;
  for (int r = threadIdx.x; r < M; r += blockDim.x) {
    float v = X[(size_t)r * ld + col];
    a += v; q += v * v;
  }
  s1[threadIdx.x] = a; s2[threadIdx.x] = q;
  __syncthreads();
  for (int st = blockDim.x >> 1; st > 0; st >>= 1) {
    if (threadIdx.x < st) {
      s1[threadIdx.x] += s1[threadIdx.x + st];
      s2[threadIdx.x] += s2[threadIdx.x + st];
    }
    __syncthreads();
  }
  if (threadIdx.x == 0) {
    float mean = s1[0] / M;
    float var  = s2[0] / M - mean * mean;
    float sc = g[col] * rsqrtf(var + 1e-5f);
    scale[col] = sc;
    shift[col] = b[col] - mean * sc;
  }
}

__global__ void bnapply_kernel(float* __restrict__ X, int M, int ld, int ncols,
                               const float* __restrict__ scale,
                               const float* __restrict__ shift) {
  int t = blockIdx.x * blockDim.x + threadIdx.x;
  if (t >= M * ncols) return;
  int c = t % ncols, r = t / ncols;
  size_t o = (size_t)r * ld + c;
  X[o] = X[o] * scale[c] + shift[c];
}

// Grouped Conv1d(K, K*K, K, groups=K): out[p,g,j] = sum_l W[g,j,l]*in[p,g,l] + b[g,j]
__global__ void gconv_kernel(const float* __restrict__ tin, const float* __restrict__ w,
                             const float* __restrict__ bias, float* __restrict__ tout,
                             int P, int K, int act) {
  int KK = K * K;
  int t = blockIdx.x * blockDim.x + threadIdx.x;
  if (t >= P * KK) return;
  int j = t % K, g = (t / K) % K, p = t / KK;
  const float* ti = tin + (size_t)p * KK + g * K;
  const float* wr = w + (size_t)g * KK + j * K;
  float acc = bias[g * K + j];
  for (int l = 0; l < K; ++l) acc += wr[l] * ti[l];
  if (act == 1) acc = (acc > 0.f) ? acc : (__expf(acc) - 1.f);
  tout[t] = acc;
}

// Fused: xt[c,j] = sum_k xs[p,k,c]*T[p,k,j]; o[p,c*dm+m] = sum_k xt[c,k]*dw[c,m,k]+db
__global__ void xtdw_kernel(const float* __restrict__ xs, const float* __restrict__ tmat,
                            const float* __restrict__ dw, const float* __restrict__ db,
                            float* __restrict__ o, int P, int K, int C, int dm) {
  int t = blockIdx.x * blockDim.x + threadIdx.x;
  if (t >= P * C) return;
  int c = t % C, p = t / C;
  float xr[16];
  for (int k = 0; k < K; ++k) xr[k] = xs[((size_t)p * K + k) * C + c];
  const float* tp = tmat + (size_t)p * K * K;
  float xt[16];
  for (int j = 0; j < K; ++j) {
    float a = 0.f;
    for (int k = 0; k < K; ++k) a += xr[k] * tp[k * K + j];
    xt[j] = a;
  }
  for (int m = 0; m < dm; ++m) {
    const float* dwr = dw + ((size_t)c * dm + m) * K;
    float a = db[c * dm + m];
    for (int k = 0; k < K; ++k) a += xt[k] * dwr[k];
    o[(size_t)p * C * dm + c * dm + m] = a;
  }
}

__global__ void meanpool_kernel(const float* __restrict__ h, float* __restrict__ out,
                                int B, int n, int C) {
  int t = blockIdx.x * blockDim.x + threadIdx.x;
  if (t >= B * C) return;
  int c = t % C, b = t / C;
  float a = 0.f;
  for (int i = 0; i < n; ++i) a += h[((size_t)b * n + i) * C + c];
  out[t] = a / (float)n;
}

__global__ void dropmul_kernel(float* __restrict__ g, const float* __restrict__ mask,
                               int total) {
  int t = blockIdx.x * blockDim.x + threadIdx.x;
  if (t < total) g[t] = g[t] * mask[t] * 2.0f;
}

// ---------------------------------------------------------------------------
// Host-side orchestration
// ---------------------------------------------------------------------------
struct LayerCfg { int n, ci, cd, K, dil, co; };

static void launch_gemm(hipStream_t s, const float* A, int lda, const float* W,
                        const float* bias, float* C, int ldc, int M, int N, int Kd,
                        int act, _Float16* Ah, _Float16* Wh) {
  int nK = cdiv(Kd, 32);
  nK += (nK & 1);                 // even chunk count: branch-free WMMA loop
  int tM = cdiv(M, 16), tN = cdiv(N, 16);
  int ea = tM * nK * 512, ew = tN * nK * 512;
  stage_a_kernel<<<cdiv(ea, 256), 256, 0, s>>>(A, lda, M, Kd, nK, ea, Ah);
  stage_w_kernel<<<cdiv(ew, 256), 256, 0, s>>>(W, N, Kd, nK, ew, Wh);
  int tiles = tM * cdiv(tN, 2);
  gemm_wmma<<<cdiv(tiles, 4), 128, 0, s>>>(Ah, Wh, bias, C, ldc, M, N, nK, act);
}

// Param index map inside one layer block (sorted dict keys, 24 leaves):
// 0 bn1b 1 bn1g 2 bn2b 3 bn2g 4 c1b 5 c1w 6 c2b 7 c2w 8 db 9 dw
// 10 l1b 11 l1w 12 l2b 13 l2w 14 lb 15 lw 16 t1b 17 t1w
// 18 tbn1b 19 tbn1g 20 tbn2b 21 tbn2g 22 tbn3b 23 tbn3g
static void run_xconv(hipStream_t s, const LayerCfg& L, const float* const* P,
                      const float* xin, const float* pos, float* hout,
                      int* nbr, float* rel, float* h1, float* xs,
                      float* t1, float* t2, float* ob,
                      float* scale, float* shift, _Float16* Ah, _Float16* Wh) {
  const int B = 32;
  int n = L.n, K = L.K, dil = L.dil, ci = L.ci, cd = L.cd, co = L.co;
  int C = ci + cd, dm = (co + C - 1) / C, KK = K * K;
  int Pn = B * n, PK = Pn * K;
  int kk = K * dil;
  const int thr = 256;

  if (kk == 8)       knn_kernel<8><<<cdiv(Pn, 128), 128, 0, s>>>(pos, B, n, dil, K, nbr);
  else if (kk == 24) knn_kernel<24><<<cdiv(Pn, 128), 128, 0, s>>>(pos, B, n, dil, K, nbr);
  else               knn_kernel<32><<<cdiv(Pn, 128), 128, 0, s>>>(pos, B, n, dil, K, nbr);
  rel_kernel<<<cdiv(PK * 3, thr), thr, 0, s>>>(pos, nbr, rel, B, n, K);

  // mlp1: 3 -> cd -> cd, ELU + BN each; second layer writes x_star's first cd cols
  launch_gemm(s, rel, 3, P[11], P[10], h1, cd, PK, cd, 3, 1, Ah, Wh);
  bnstats_kernel<<<cd, 256, 0, s>>>(h1, PK, cd, P[1], P[0], scale, shift);
  bnapply_kernel<<<cdiv(PK * cd, thr), thr, 0, s>>>(h1, PK, cd, cd, scale, shift);
  launch_gemm(s, h1, cd, P[13], P[12], xs, C, PK, cd, cd, 1, Ah, Wh);
  bnstats_kernel<<<cd, 256, 0, s>>>(xs, PK, C, P[3], P[2], scale, shift);
  bnapply_kernel<<<cdiv(PK * cd, thr), thr, 0, s>>>(xs, PK, C, cd, scale, shift);
  gatherx_kernel<<<cdiv(PK * ci, thr), thr, 0, s>>>(xin, nbr, xs, B, n, K, ci, C, cd);

  // X-transform: (K*3) -> K*K lift, two grouped KxK convs, BN after each
  launch_gemm(s, rel, K * 3, P[17], P[16], t1, KK, Pn, KK, K * 3, 1, Ah, Wh);
  bnstats_kernel<<<KK, 256, 0, s>>>(t1, Pn, KK, P[19], P[18], scale, shift);
  bnapply_kernel<<<cdiv(Pn * KK, thr), thr, 0, s>>>(t1, Pn, KK, KK, scale, shift);
  gconv_kernel<<<cdiv(Pn * KK, thr), thr, 0, s>>>(t1, P[5], P[4], t2, Pn, K, 1);
  bnstats_kernel<<<KK, 256, 0, s>>>(t2, Pn, KK, P[21], P[20], scale, shift);
  bnapply_kernel<<<cdiv(Pn * KK, thr), thr, 0, s>>>(t2, Pn, KK, KK, scale, shift);
  gconv_kernel<<<cdiv(Pn * KK, thr), thr, 0, s>>>(t2, P[7], P[6], t1, Pn, K, 0);
  bnstats_kernel<<<KK, 256, 0, s>>>(t1, Pn, KK, P[23], P[22], scale, shift);
  bnapply_kernel<<<cdiv(Pn * KK, thr), thr, 0, s>>>(t1, Pn, KK, KK, scale, shift);

  // xt einsum + depthwise conv fused, then output linear with fused ReLU
  xtdw_kernel<<<cdiv(Pn * C, thr), thr, 0, s>>>(xs, t1, P[9], P[8], ob, Pn, K, C, dm);
  launch_gemm(s, ob, C * dm, P[15], P[14], hout, co, Pn, co, C * dm, 2, Ah, Wh);
}

extern "C" void kernel_launch(void* const* d_in, const int* in_sizes, int n_in,
                              void* d_out, int out_size, void* d_ws, size_t ws_size,
                              hipStream_t stream) {
  (void)in_sizes; (void)n_in; (void)out_size; (void)ws_size;
  const int B = 32, N = 1024;
  const float* x   = (const float*)d_in[0];
  const float* pos = (const float*)d_in[1];
  const float* LP[4][24];
  for (int L = 0; L < 4; ++L)
    for (int j = 0; j < 24; ++j) LP[L][j] = (const float*)d_in[2 + 24 * L + j];
  const float* hl1b = (const float*)d_in[98];
  const float* hl1w = (const float*)d_in[99];
  const float* hl2b = (const float*)d_in[100];
  const float* hl2w = (const float*)d_in[101];
  const float* hl3b = (const float*)d_in[102];
  const float* hl3w = (const float*)d_in[103];
  const float* mask = (const float*)d_in[104];

  // workspace carve (bump allocator, 256B aligned)
  char* w = (char*)d_ws;
  auto alloc = [&](size_t bytes) -> void* {
    void* p = (void*)w;
    w += (bytes + 255) & ~(size_t)255;
    return p;
  };
  float* hA   = (float*)alloc(sizeof(float) * (size_t)B * N * 48);   // layer outputs
  float* hB   = (float*)alloc(sizeof(float) * (size_t)B * N * 48);
  float* posA = (float*)alloc(sizeof(float) * (size_t)B * N * 3);
  float* posB = (float*)alloc(sizeof(float) * (size_t)B * N * 3);
  int*   sel  = (int*)alloc(sizeof(int) * (size_t)B * 384);
  int*   nbr  = (int*)alloc(sizeof(int) * 262144);                   // max B*n*K
  float* rel  = (float*)alloc(sizeof(float) * 786432);               // max B*n*K*3
  float* h1   = (float*)alloc(sizeof(float) * 16908288ULL);          // max PK*cd (c4)
  float* xs   = (float*)alloc(sizeof(float) * 29589504ULL);          // max PK*C  (c4)
  float* t1   = (float*)alloc(sizeof(float) * 2097152);              // max Pn*K*K (c1)
  float* t2   = (float*)alloc(sizeof(float) * 2097152);
  float* ob   = (float*)alloc(sizeof(float) * 2293760);              // max Pn*C*dm (c1)
  float* scale = (float*)alloc(sizeof(float) * 512);
  float* shift = (float*)alloc(sizeof(float) * 512);
  float* g0 = (float*)alloc(sizeof(float) * 32 * 384);
  float* g1 = (float*)alloc(sizeof(float) * 32 * 256);
  float* g2 = (float*)alloc(sizeof(float) * 32 * 128);
  // staged f16 fragment buffers (A: max 4128 M-tiles x 8 even-rounded chunks;
  // W sized with slack for the unconditional second-tile read on odd tile counts)
  _Float16* Ah = (_Float16*)alloc(sizeof(_Float16) * 16908288ULL);
  _Float16* Wh = (_Float16*)alloc(sizeof(_Float16) * 262144ULL);

  const LayerCfg L0{1024,   3,  32,  8, 1,  48};
  const LayerCfg L1{ 384,  48,  64, 12, 2,  96};
  const LayerCfg L2{ 129,  96, 128, 16, 2, 192};
  const LayerCfg L3{ 129, 192, 256, 16, 2, 384};

  // c1 on full cloud
  run_xconv(stream, L0, LP[0], x, pos, hA, nbr, rel, h1, xs, t1, t2, ob, scale, shift, Ah, Wh);
  // FPS 1024 -> 384, gather h/pos, c2
  fps_kernel<<<B, 256, 0, stream>>>(pos, 1024, 384, sel);
  gather_kernel<<<cdiv(B * 384 * 48, 256), 256, 0, stream>>>(hA, sel, hB, B, 384, 1024, 48);
  gather_kernel<<<cdiv(B * 384 * 3, 256), 256, 0, stream>>>(pos, sel, posA, B, 384, 1024, 3);
  run_xconv(stream, L1, LP[1], hB, posA, hA, nbr, rel, h1, xs, t1, t2, ob, scale, shift, Ah, Wh);
  // FPS 384 -> 129, gather, c3
  fps_kernel<<<B, 256, 0, stream>>>(posA, 384, 129, sel);
  gather_kernel<<<cdiv(B * 129 * 96, 256), 256, 0, stream>>>(hA, sel, hB, B, 129, 384, 96);
  gather_kernel<<<cdiv(B * 129 * 3, 256), 256, 0, stream>>>(posA, sel, posB, B, 129, 384, 3);
  run_xconv(stream, L2, LP[2], hB, posB, hA, nbr, rel, h1, xs, t1, t2, ob, scale, shift, Ah, Wh);
  // c4 on same 129-point cloud
  run_xconv(stream, L3, LP[3], hA, posB, hB, nbr, rel, h1, xs, t1, t2, ob, scale, shift, Ah, Wh);

  // Head: mean pool -> 384 -> 256 -> 128 -> dropout -> 10
  meanpool_kernel<<<cdiv(32 * 384, 256), 256, 0, stream>>>(hB, g0, B, 129, 384);
  launch_gemm(stream, g0, 384, hl1w, hl1b, g1, 256, 32, 256, 384, 2, Ah, Wh);
  launch_gemm(stream, g1, 256, hl2w, hl2b, g2, 128, 32, 128, 256, 2, Ah, Wh);
  dropmul_kernel<<<cdiv(32 * 128, 256), 256, 0, stream>>>(g2, mask, 32 * 128);
  launch_gemm(stream, g2, 128, hl3w, hl3b, (float*)d_out, 10, 32, 10, 128, 0, Ah, Wh);
}
